// rnn_14328010900226
// MI455X (gfx1250) — matrix operand order repaired and checked by reading, not compiled, not run on hardware
//
#include <hip/hip_runtime.h>

#define SEQ    512
#define BATCH  4096
#define IN_DIM 64
#define HID    128
#define BN     32            // batch rows per workgroup
#define NT     (BN / 16)     // n-tiles per wave
#define WAVES  8             // 8 waves * 16 rows = 128 = HID output features
#define THREADS (WAVES * 32)

typedef __attribute__((ext_vector_type(16))) __bf16       v16bf;
typedef __attribute__((ext_vector_type(8)))  __bf16       v8bf;
typedef __attribute__((ext_vector_type(8)))  float        v8f;
typedef __attribute__((ext_vector_type(4)))  float        v4f;
typedef __attribute__((ext_vector_type(4)))  unsigned int v4u;

struct U32x8 { v4u lo, hi; };

// LDS layout (dynamic shared):
//   raw x : 4-deep ring of fp32 tiles (async-copy targets, prefetch dist 2)
//   xbf   : double-buffered bf16 x tiles [n][k] (converted one step ahead)
//   h     : double-buffered bf16 h^T tiles [n][m]
#define XRAWB  (BN * IN_DIM * 4)   // 8192
#define XBFB   (BN * IN_DIM * 2)   // 4096
#define HBYTES (BN * HID * 2)      // 8192
#define XRAW(i) ((i) * XRAWB)                          // 0..32K
#define XBF(i)  (4 * XRAWB + (i) * XBFB)               // 32K..40K
#define HBUF(i) (4 * XRAWB + 2 * XBFB + (i) * HBYTES)  // 40K..56K
#define SMEM_BYTES (4 * XRAWB + 2 * XBFB + 2 * HBYTES) // 57344

__device__ __forceinline__ float tanh_fast(float x) {
#if __has_builtin(__builtin_amdgcn_tanhf)
  return __builtin_amdgcn_tanhf(x);           // v_tanh_f32 (gfx1250)
#else
  float e = __builtin_amdgcn_exp2f(2.885390081777927f * x);  // e^(2x)
  return 1.0f - 2.0f / (e + 1.0f);
#endif
}

// A-fragment (16x32 bf16, weights): lane l holds row m = m0 + (l&15);
// lanes 0-15: K = k0h..k0h+15 (k0h = 32*s),
// lanes 16-31: same rows, K shifted by 16 (k0h = 32*s + 16).
__device__ __forceinline__ v16bf load_afrag(const float* __restrict__ wrow, int k0h) {
  float f[16];
  *(v4f*)&f[0]  = *(const v4f*)(wrow + k0h);
  *(v4f*)&f[4]  = *(const v4f*)(wrow + k0h + 4);
  *(v4f*)&f[8]  = *(const v4f*)(wrow + k0h + 8);
  *(v4f*)&f[12] = *(const v4f*)(wrow + k0h + 12);
  v16bf a;
#pragma unroll
  for (int i = 0; i < 16; ++i) a[i] = (__bf16)f[i];
  return a;
}

// Forced (un-sinkable) B-fragment load: two ds_load_b128 into caller regs.
__device__ __forceinline__ void ds_load_frag(unsigned addr, v4u& lo, v4u& hi) {
  asm volatile("ds_load_b128 %0, %2\n\t"
               "ds_load_b128 %1, %2 offset:16"
               : "=v"(lo), "=v"(hi)
               : "v"(addr)
               : "memory");
}

// Gate the consumption of a fragment on DScnt <= n (ties via "+v" force
// load -> wait -> WMMA ordering while later loads stay in flight).
__device__ __forceinline__ void ds_wait_frag(v4u& lo, v4u& hi, int n) {
  asm volatile("s_wait_dscnt %2" : "+v"(lo), "+v"(hi) : "i"(n) : "memory");
}

// Async-copy one x_t slice (BN*IN_DIM fp32 = 8 KB, contiguous) into LDS.
// 256 threads * 16 B * 2 instructions = 8 KB. Tracked by ASYNCcnt.
__device__ __forceinline__ void prefetch_x(const char* smem, int xbase,
                                           const float* __restrict__ xg, int tid) {
  unsigned o0   = (unsigned)(tid * 16);
  unsigned lds0 = (unsigned)(uintptr_t)(smem + xbase) + o0;
  const char* g0 = (const char*)xg + o0;
  asm volatile("global_load_async_to_lds_b128 %0, %1, off"
               :: "v"(lds0), "v"(g0) : "memory");
  unsigned lds1 = lds0 + THREADS * 16;
  const char* g1 = g0 + THREADS * 16;
  asm volatile("global_load_async_to_lds_b128 %0, %1, off"
               :: "v"(lds1), "v"(g1) : "memory");
}

// Cooperative fp32 -> bf16 conversion of one x tile (2048 elements):
// each of the 256 threads converts 8 elements (2 b128 loads, 1 b128 store).
__device__ __forceinline__ void convert_x(char* smem, int raw_base, int bf_base,
                                          int tid) {
  const v4f* src = (const v4f*)(smem + raw_base + tid * 32);
  v4f f0 = src[0], f1 = src[1];
  v8bf b;
#pragma unroll
  for (int i = 0; i < 4; ++i) b[i] = (__bf16)f0[i];
#pragma unroll
  for (int i = 0; i < 4; ++i) b[4 + i] = (__bf16)f1[i];
  *(v4u*)(smem + bf_base + tid * 16) = __builtin_bit_cast(v4u, b);
}

__global__ __launch_bounds__(THREADS, 1)
void rnn_wmma_kernel(const float* __restrict__ x,
                     const float* __restrict__ W_ih,
                     const float* __restrict__ b_ih,
                     const float* __restrict__ W_hh,
                     const float* __restrict__ b_hh,
                     float* __restrict__ out) {
  extern __shared__ char smem[];
  const int tid  = threadIdx.x;
  const int lane = tid & 31;
  const int wave = tid >> 5;
  const int hf   = lane >> 4;   // half-wave id
  const int l16  = lane & 15;
  const int wgb0 = blockIdx.x * BN;

  // h0 = 0: zero the first h buffer (8 KB, 32 B per thread)
#pragma unroll
  for (int i = 0; i < 2; ++i) {
    v4u z = {0u, 0u, 0u, 0u};
    *(v4u*)(smem + HBUF(0) + (tid + i * THREADS) * 16) = z;
  }

  // Load this wave's static A-fragments (weights) into registers, once.
  // m-strip m0..m0+15; combined K = [0,64) from W_ih, [64,192) from W_hh.
  const int m0   = wave * 16;
  const int mrow = m0 + l16;
  v16bf afrag[6];
#pragma unroll
  for (int s = 0; s < 2; ++s)
    afrag[s] = load_afrag(W_ih + (size_t)mrow * IN_DIM, 32 * s + 16 * hf);
#pragma unroll
  for (int s = 0; s < 4; ++s)
    afrag[2 + s] = load_afrag(W_hh + (size_t)mrow * HID, 32 * s + 16 * hf);

  // Bias in C/D layout: lane l, acc element r -> feature m = m0 + 8*hf + r.
  v8f bias;
#pragma unroll
  for (int r = 0; r < 8; ++r) {
    int m = m0 + 8 * hf + r;
    bias[r] = b_ih[m] + b_hh[m];
  }

  // Per-lane LDS byte offsets for this wave's B-fragments (step-invariant
  // up to the double-buffer select).
  const unsigned smembase = (unsigned)(uintptr_t)smem;
  unsigned xoff[NT], hoff[NT];
#pragma unroll
  for (int nt = 0; nt < NT; ++nt) {
    const int n = nt * 16 + l16;
    xoff[nt] = smembase + (unsigned)(n * (IN_DIM * 2) + (16 * hf) * 2);
    hoff[nt] = smembase + (unsigned)(n * (HID * 2) + (16 * hf) * 2);
  }

  const float* xg_base = x + (size_t)wgb0 * IN_DIM;
  // Prime the pipeline: t=0 and t=1 in flight, convert x[0] to bf16.
  prefetch_x(smem, XRAW(0), xg_base, tid);
  prefetch_x(smem, XRAW(1), xg_base + (size_t)BATCH * IN_DIM, tid);
  asm volatile("s_wait_asynccnt 2" ::: "memory");  // x[0] landed
  convert_x(smem, XRAW(0), XBF(0), tid);

#pragma unroll 4
  for (int t = 0; t < SEQ; ++t) {
    const int p = t & 1;

    // Prefetch raw x[t+2] into ring slot (t+2)&3 == (t-2)&3: its last
    // readers finished by the barrier of step t-1, so issuing before this
    // step's barrier is safe -> only ONE barrier per step.
    if (t < SEQ - 2) {
      prefetch_x(smem, XRAW((t + 2) & 3),
                 xg_base + (size_t)(t + 2) * BATCH * IN_DIM, tid);
      asm volatile("s_wait_asynccnt 2" ::: "memory");  // x[t+1] landed (in-order)
    } else {
      asm volatile("s_wait_asynccnt 0" ::: "memory");
    }

    // Single per-step barrier: publishes every wave's async x copies and the
    // cooperative x[t] conversion, and separates step t-1 / t compute.
    // (Also drains DScnt to 0 -> manual thresholds below are exact.)
    __syncthreads();

    // Issue ALL 24 ds_load_b128 back-to-back via asm (scheduler cannot sink
    // or coalesce them), then consume with descending s_wait_dscnt: only the
    // first fragment's LDS latency is exposed per step.
    v4u flo[6][NT], fhi[6][NT];
#pragma unroll
    for (int s = 0; s < 6; ++s) {
#pragma unroll
      for (int nt = 0; nt < NT; ++nt) {
        unsigned addr = (s < 2)
                            ? xoff[nt] + (unsigned)(XBF(p) + (32 * s) * 2)
                            : hoff[nt] + (unsigned)(HBUF(p) + (32 * (s - 2)) * 2);
        ds_load_frag(addr, flo[s][nt], fhi[s][nt]);
      }
    }

    v8f acc[NT];
#pragma unroll
    for (int nt = 0; nt < NT; ++nt) acc[nt] = bias;

#pragma unroll
    for (int s = 0; s < 6; ++s) {
#pragma unroll
      for (int nt = 0; nt < NT; ++nt) {
        const int k = s * NT + nt;          // fragment index 0..11
        ds_wait_frag(flo[s][nt], fhi[s][nt], 22 - 2 * k);
        U32x8 u; u.lo = flo[s][nt]; u.hi = fhi[s][nt];
        v16bf b = __builtin_bit_cast(v16bf, u);
        acc[nt] = __builtin_amdgcn_wmma_f32_16x16x32_bf16(
            false, afrag[s], false, b, (short)0, acc[nt], false, false);
      }
    }

    // Convert raw x[t+1] -> bf16 (consumed next step). Independent DS/VALU
    // work that fills the WMMA drain latency. DScnt is fully drained above
    // (last ds_wait_frag threshold is 0), so compiler wait accounting stays
    // exact for its own DS ops below.
    if (t < SEQ - 1)
      convert_x(smem, XRAW((t + 1) & 3), XBF((t + 1) & 1), tid);

    // tanh, pack to bf16, store h_{t+1}^T ([n][m], 8 m-contiguous values
    // per lane -> single ds_store_b128 per n-tile).
#pragma unroll
    for (int nt = 0; nt < NT; ++nt) {
      const int n = nt * 16 + l16;
      float hv[8];
#pragma unroll
      for (int r = 0; r < 8; ++r) hv[r] = tanh_fast(acc[nt][r]);

      v8bf hb;
#pragma unroll
      for (int r = 0; r < 8; ++r) hb[r] = (__bf16)hv[r];
      char* hp = smem + HBUF(1 - p) + n * (HID * 2) + (m0 + 8 * hf) * 2;
      *(v4u*)hp = __builtin_bit_cast(v4u, hb);

      if (t == SEQ - 1) {  // final h in fp32 straight to output (2x b128)
        float* op = out + (size_t)(wgb0 + n) * HID + m0 + 8 * hf;
        v4f o0, o1;
#pragma unroll
        for (int r = 0; r < 4; ++r) { o0[r] = hv[r]; o1[r] = hv[4 + r]; }
        *(v4f*)op = o0;
        *(v4f*)(op + 4) = o1;
      }
    }
  }
}

extern "C" void kernel_launch(void* const* d_in, const int* in_sizes, int n_in,
                              void* d_out, int out_size, void* d_ws, size_t ws_size,
                              hipStream_t stream) {
  const float* x    = (const float*)d_in[0];
  const float* W_ih = (const float*)d_in[1];
  const float* b_ih = (const float*)d_in[2];
  const float* W_hh = (const float*)d_in[3];
  const float* b_hh = (const float*)d_in[4];
  float* out = (float*)d_out;

  dim3 grid(BATCH / BN);   // 128 workgroups
  dim3 block(THREADS);     // 256 threads = 8 wave32
  hipLaunchKernelGGL(rnn_wmma_kernel, grid, block, SMEM_BYTES, stream,
                     x, W_ih, b_ih, W_hh, b_hh, out);
}
